// TrueSlidingWindowEncoder_61795989455015
// MI455X (gfx1250) — matrix operand order repaired and checked by reading, not compiled, not run on hardware
//
#include <hip/hip_runtime.h>

// ---------------- problem constants ----------------
#define BB   2
#define SS   2048
#define DD   768
#define HH   12
#define DHH  64
#define LL   2
#define DFFN 3072
#define MM   (BB*SS)          // 4096 rows
#define WINH 256              // WIN/2
#define NKT  34               // key tiles per query tile (544 keys)
#define COLS 544
#define EPSL 1e-5f

typedef __attribute__((ext_vector_type(16))) __bf16 v16bf;
typedef __attribute__((ext_vector_type(8)))  float  v8f;

union FragB {
    uint4          q[2];
    unsigned short h[16];
    v16bf          v;
};

static __device__ __forceinline__ unsigned short f2bf(float f) {
    union { float f; unsigned u; } v; v.f = f;
    unsigned u = v.u;
    u += 0x7fffu + ((u >> 16) & 1u);        // round-to-nearest-even
    return (unsigned short)(u >> 16);
}

static __device__ __forceinline__ v8f wmma_bf16(const FragB& a, const FragB& b, v8f c) {
    return __builtin_amdgcn_wmma_f32_16x16x32_bf16(
        false, a.v, false, b.v, (short)0, c, false, false);
}

// ---------------- elementwise converts ----------------
__global__ __launch_bounds__(256) void cvt_bf16_kernel(const float* __restrict__ src,
                                                       unsigned short* __restrict__ dst, int n) {
    int i = blockIdx.x * 256 + threadIdx.x;
    if (i < n) dst[i] = f2bf(src[i]);
}

__global__ __launch_bounds__(256) void init_x_kernel(const float* __restrict__ src,
                                                     float* __restrict__ x,
                                                     unsigned short* __restrict__ xbf, int n) {
    int i = blockIdx.x * 256 + threadIdx.x;
    if (i < n) { float v = src[i]; x[i] = v; xbf[i] = f2bf(v); }
}

// ---------------- tiled bf16 GEMM:  C[M,N] = A[M,K] * W[N,K]^T + bias ----------------
// 256 threads = 8 waves (4 in M x 2 in N).  WG tile 128x128, wave tile 32x64
// (2x4 wmma frags).  Double-buffered LDS -> one barrier per K step.
__global__ __launch_bounds__(256) void gemm_bf16_kernel(
        const unsigned short* __restrict__ A,   // M x K  bf16 row-major
        const unsigned short* __restrict__ W,   // N x K  bf16 row-major
        const float*          __restrict__ bias,// N
        float*                __restrict__ Cf,  // optional f32 out
        unsigned short*       __restrict__ Cbf, // optional bf16 out
        int M, int N, int K, int relu) {
    __shared__ unsigned short As[2][128 * 40]; // row stride 40 elem (80B): conflict-free b128
    __shared__ unsigned short Bs[2][128 * 40];

    const int t    = threadIdx.x;
    const int m0   = blockIdx.y * 128;
    const int n0   = blockIdx.x * 128;
    const int wid  = t >> 5, lane = t & 31;
    const int wm   = wid >> 1, wn = wid & 1;   // 4 x 2 wave grid
    const int lrow = lane & 15, lhi = lane >> 4;

    const int sr = t >> 2;                     // staging row 0..63 (x2 iters -> 0..127)
    const int sc = (t & 3) * 8;                // staging col offset (elements)

    v8f acc[2][4] = {};

    int buf = 0;
    for (int kk = 0; kk < K; kk += 32, buf ^= 1) {
        // ---- stage A (128x32) and B (128x32) tiles into buffer `buf` ----
        #pragma unroll
        for (int i = 0; i < 2; ++i) {
            int r = sr + i * 64;
            *(uint4*)&As[buf][r * 40 + sc] =
                *(const uint4*)&A[(size_t)(m0 + r) * K + kk + sc];
            *(uint4*)&Bs[buf][r * 40 + sc] =
                *(const uint4*)&W[(size_t)(n0 + r) * K + kk + sc];
        }
        // prefetch tile kk+64 into caches (global_prefetch_b8)
        if (kk + 64 < K) {
            __builtin_prefetch(&A[(size_t)(m0 + sr) * K + kk + 64 + sc], 0, 1);
            __builtin_prefetch(&W[(size_t)(n0 + sr) * K + kk + 64 + sc], 0, 1);
        }
        __syncthreads();

        // ---- fragments (ISA A/B VGPR layouts) ----
        FragB af[2], bfr[4];
        #pragma unroll
        for (int mt = 0; mt < 2; ++mt) {
            const unsigned short* p = &As[buf][(wm * 32 + mt * 16 + lrow) * 40 + lhi * 8];
            af[mt].q[0] = *(const uint4*)p;          // K = lhi*8 + 0..7
            af[mt].q[1] = *(const uint4*)(p + 16);   // K = 16 + lhi*8 + 0..7
        }
        #pragma unroll
        for (int nt = 0; nt < 4; ++nt) {
            const unsigned short* p = &Bs[buf][(wn * 64 + nt * 16 + lrow) * 40 + lhi * 8];
            bfr[nt].q[0] = *(const uint4*)p;         // K = lhi*8 + 0..7
            bfr[nt].q[1] = *(const uint4*)(p + 16);  // K = 16 + lhi*8 + 0..7
        }
        #pragma unroll
        for (int mt = 0; mt < 2; ++mt)
            #pragma unroll
            for (int nt = 0; nt < 4; ++nt)
                acc[mt][nt] = wmma_bf16(af[mt], bfr[nt], acc[mt][nt]);
        // no trailing barrier: next iteration writes the other buffer
    }

    // ---- epilogue: bias (+relu), f32 and/or bf16 stores ----
    #pragma unroll
    for (int mt = 0; mt < 2; ++mt)
        #pragma unroll
        for (int nt = 0; nt < 4; ++nt) {
            int col = n0 + wn * 64 + nt * 16 + lrow;
            float bv = bias ? bias[col] : 0.f;
            #pragma unroll
            for (int r = 0; r < 8; ++r) {
                int row = m0 + wm * 32 + mt * 16 + r + lhi * 8;
                float v = acc[mt][nt][r] + bv;
                if (relu) v = v > 0.f ? v : 0.f;
                if (Cf)  Cf [(size_t)row * N + col] = v;
                if (Cbf) Cbf[(size_t)row * N + col] = f2bf(v);
            }
        }
}

// ---------------- banded attention: one wave per (b,h,16-query tile) ----------------
// dyn LDS per wave: 16x544 f32 score band + 32x72 bf16 V staging
__global__ __launch_bounds__(128) void attn_kernel(
        const unsigned short* __restrict__ Q,   // (b*S+s)*D + h*64 + dh, bf16
        const unsigned short* __restrict__ Km,
        const unsigned short* __restrict__ V,
        unsigned short*       __restrict__ ctx) {
    extern __shared__ char smem[];
    const int wid  = threadIdx.x >> 5;
    const int lane = threadIdx.x & 31;
    float*          sb = (float*)smem + (size_t)wid * (16 * COLS);
    unsigned short* vb = (unsigned short*)(smem + 4 * 16 * COLS * 4) + (size_t)wid * (32 * 72);

    const int tile = blockIdx.x * 4 + wid;      // 3072 tiles total
    const int qt   = tile & 127;
    const int bh   = tile >> 7;
    const int h    = bh % HH;
    const int b    = bh / HH;
    const int q0   = qt * 16;
    const int lrow = lane & 15, lhi = lane >> 4;
    const float scale = 0.125f;                 // DH^-0.5

    // Q fragments (16 x 64 -> two 16x32 A-frags)
    FragB aq[2];
    {
        size_t qb = ((size_t)(b * SS + q0 + lrow)) * DD + h * DHH;
        #pragma unroll
        for (int c = 0; c < 2; ++c) {
            aq[c].q[0] = *(const uint4*)&Q[qb + c * 32 + lhi * 8];
            aq[c].q[1] = *(const uint4*)&Q[qb + c * 32 + 16 + lhi * 8];
        }
    }

    // ---- pass 1: banded scores -> LDS ----
    for (int j = 0; j < NKT; ++j) {
        int kb  = q0 - WINH + j * 16;
        int key = kb + lrow;                    // this lane's key column
        FragB bk[2];
        if ((unsigned)key < (unsigned)SS) {
            size_t kbp = ((size_t)(b * SS + key)) * DD + h * DHH;
            #pragma unroll
            for (int c = 0; c < 2; ++c) {
                bk[c].q[0] = *(const uint4*)&Km[kbp + c * 32 + lhi * 8];
                bk[c].q[1] = *(const uint4*)&Km[kbp + c * 32 + 16 + lhi * 8];
            }
        } else {
            uint4 z = make_uint4(0, 0, 0, 0);
            bk[0].q[0] = bk[0].q[1] = bk[1].q[0] = bk[1].q[1] = z;
        }
        v8f s = {};
        s = wmma_bf16(aq[0], bk[0], s);
        s = wmma_bf16(aq[1], bk[1], s);
        #pragma unroll
        for (int r = 0; r < 8; ++r) {
            int row = q0 + r + lhi * 8;
            int dd  = row - key; if (dd < 0) dd = -dd;
            bool ok = ((unsigned)key < (unsigned)SS) && (dd <= WINH);
            float v = ok ? s[r] * scale : -1e30f;
            sb[(r + lhi * 8) * COLS + j * 16 + lrow] = v;
        }
    }

    // ---- softmax: lane handles row=lrow, half of the 544 columns ----
    float rowinv;
    {
        float* rp = sb + lrow * COLS + lhi * (COLS / 2);
        float m = -1e30f;
        for (int c = 0; c < COLS / 2; ++c) m = fmaxf(m, rp[c]);
        m = fmaxf(m, __shfl_xor(m, 16, 32));
        float sum = 0.f;
        for (int c = 0; c < COLS / 2; ++c) {
            float p = __expf(rp[c] - m);
            rp[c] = p;
            sum += p;
        }
        sum += __shfl_xor(sum, 16, 32);
        rowinv = 1.0f / sum;                    // valid in lanes lrow and lrow+16
    }

    // ---- pass 3: O = P x V over 17 key-pair (32-key) tiles ----
    v8f acc[4] = {};
    for (int jj = 0; jj < COLS / 32; ++jj) {
        int kb = q0 - WINH + jj * 32;
        // stage V pair-tile (32 keys x 64 dh) into LDS, row per lane
        {
            int key = kb + lane;
            unsigned short* dst = vb + lane * 72;
            if ((unsigned)key < (unsigned)SS) {
                const unsigned short* src = &V[((size_t)(b * SS + key)) * DD + h * DHH];
                #pragma unroll
                for (int w = 0; w < 8; ++w)
                    *(uint4*)&dst[w * 8] = *(const uint4*)&src[w * 8];
            } else {
                uint4 z = make_uint4(0, 0, 0, 0);
                #pragma unroll
                for (int w = 0; w < 8; ++w) *(uint4*)&dst[w * 8] = z;
            }
        }
        // P A-fragment from LDS scores (already exp'd), f32 -> bf16
        FragB ap;
        {
            const float* rp = sb + lrow * COLS + jj * 32 + lhi * 8;
            #pragma unroll
            for (int i = 0; i < 8; ++i) ap.h[i]     = f2bf(rp[i]);
            #pragma unroll
            for (int i = 0; i < 8; ++i) ap.h[8 + i] = f2bf(rp[16 + i]);
        }
        // V B-fragments: lane's column dh, K = keys (strided LDS reads)
        #pragma unroll
        for (int nt = 0; nt < 4; ++nt) {
            FragB bv;
            int dh = nt * 16 + lrow;
            #pragma unroll
            for (int i = 0; i < 16; ++i)
                bv.h[i] = vb[(lhi * 8 + i + (i & 8)) * 72 + dh];
            acc[nt] = wmma_bf16(ap, bv, acc[nt]);
        }
    }

    // ---- normalize by row sum and store bf16 context ----
    #pragma unroll
    for (int nt = 0; nt < 4; ++nt)
        #pragma unroll
        for (int r = 0; r < 8; ++r) {
            int   rr  = r + lhi * 8;
            float inv = __shfl(rowinv, rr, 32);
            int   qrow = q0 + rr;
            float v = acc[nt][r] * inv;
            ctx[((size_t)(b * SS + qrow)) * DD + h * DHH + nt * 16 + lrow] = f2bf(v);
        }
}

// ---------------- residual + LayerNorm (row per block, D=768, 256 thr) ----------------
__global__ __launch_bounds__(256) void ln_kernel(
        const float* __restrict__ x, const float* __restrict__ o,
        const float* __restrict__ g, const float* __restrict__ be,
        float* __restrict__ yf, unsigned short* __restrict__ ybf) {
    __shared__ float red[8];
    const int row = blockIdx.x;
    const int t   = threadIdx.x;
    const float* xp = x + (size_t)row * DD;
    const float* op = o + (size_t)row * DD;

    float v[3];
    #pragma unroll
    for (int i = 0; i < 3; ++i) v[i] = xp[t + i * 256] + op[t + i * 256];

    float s = v[0] + v[1] + v[2];
    #pragma unroll
    for (int m = 16; m >= 1; m >>= 1) s += __shfl_xor(s, m, 32);
    if ((t & 31) == 0) red[t >> 5] = s;
    __syncthreads();
    float tot = 0.f;
    #pragma unroll
    for (int i = 0; i < 8; ++i) tot += red[i];
    float mean = tot * (1.0f / (float)DD);
    __syncthreads();

    float d0 = v[0] - mean, d1 = v[1] - mean, d2 = v[2] - mean;
    float s2 = d0 * d0 + d1 * d1 + d2 * d2;
    #pragma unroll
    for (int m = 16; m >= 1; m >>= 1) s2 += __shfl_xor(s2, m, 32);
    if ((t & 31) == 0) red[t >> 5] = s2;
    __syncthreads();
    float tot2 = 0.f;
    #pragma unroll
    for (int i = 0; i < 8; ++i) tot2 += red[i];
    float rstd = rsqrtf(tot2 * (1.0f / (float)DD) + EPSL);

    #pragma unroll
    for (int i = 0; i < 3; ++i) {
        int c = t + i * 256;
        float y = (v[i] - mean) * rstd * g[c] + be[c];
        yf [(size_t)row * DD + c] = y;
        ybf[(size_t)row * DD + c] = f2bf(y);
    }
}

// ---------------- host orchestration ----------------
static inline size_t alignup(size_t v) { return (v + 255) & ~(size_t)255; }

extern "C" void kernel_launch(void* const* d_in, const int* in_sizes, int n_in,
                              void* d_out, int out_size, void* d_ws, size_t ws_size,
                              hipStream_t stream) {
    (void)in_sizes; (void)n_in; (void)out_size; (void)ws_size;

    const float* src = (const float*)d_in[0];
    const float* qw  = (const float*)d_in[1];
    const float* qb  = (const float*)d_in[2];
    const float* kw  = (const float*)d_in[3];
    const float* kb  = (const float*)d_in[4];
    const float* vw  = (const float*)d_in[5];
    const float* vb  = (const float*)d_in[6];
    const float* ow  = (const float*)d_in[7];
    const float* ob  = (const float*)d_in[8];
    const float* w1  = (const float*)d_in[9];
    const float* b1  = (const float*)d_in[10];
    const float* w2  = (const float*)d_in[11];
    const float* b2  = (const float*)d_in[12];
    const float* g1  = (const float*)d_in[13];
    const float* be1 = (const float*)d_in[14];
    const float* g2  = (const float*)d_in[15];
    const float* be2 = (const float*)d_in[16];

    char* ws = (char*)d_ws;
    size_t off = 0;
    auto carve = [&](size_t bytes) -> char* {
        char* p = ws + off;
        off = alignup(off + bytes);
        return p;
    };

    float*          x     = (float*)         carve((size_t)MM * DD * 4);
    unsigned short* xbf   = (unsigned short*)carve((size_t)MM * DD * 2);
    unsigned short* qbf   = (unsigned short*)carve((size_t)MM * DD * 2);
    unsigned short* kbf   = (unsigned short*)carve((size_t)MM * DD * 2);
    unsigned short* vbf   = (unsigned short*)carve((size_t)MM * DD * 2);
    unsigned short* ctx   = (unsigned short*)carve((size_t)MM * DD * 2);
    float*          proj  = (float*)         carve((size_t)MM * DD * 4);
    unsigned short* ffnbf = (unsigned short*)carve((size_t)MM * DFFN * 2);
    unsigned short* qwb   = (unsigned short*)carve((size_t)LL * DD * DD * 2);
    unsigned short* kwb   = (unsigned short*)carve((size_t)LL * DD * DD * 2);
    unsigned short* vwb   = (unsigned short*)carve((size_t)LL * DD * DD * 2);
    unsigned short* owb   = (unsigned short*)carve((size_t)LL * DD * DD * 2);
    unsigned short* w1b   = (unsigned short*)carve((size_t)LL * DFFN * DD * 2);
    unsigned short* w2b   = (unsigned short*)carve((size_t)LL * DD * DFFN * 2);

    // weight/src conversions
    {
        int nw = LL * DD * DD;
        int nf = LL * DFFN * DD;
        cvt_bf16_kernel<<<(nw + 255) / 256, 256, 0, stream>>>(qw, qwb, nw);
        cvt_bf16_kernel<<<(nw + 255) / 256, 256, 0, stream>>>(kw, kwb, nw);
        cvt_bf16_kernel<<<(nw + 255) / 256, 256, 0, stream>>>(vw, vwb, nw);
        cvt_bf16_kernel<<<(nw + 255) / 256, 256, 0, stream>>>(ow, owb, nw);
        cvt_bf16_kernel<<<(nf + 255) / 256, 256, 0, stream>>>(w1, w1b, nf);
        cvt_bf16_kernel<<<(nf + 255) / 256, 256, 0, stream>>>(w2, w2b, nf);
        int nx = MM * DD;
        init_x_kernel<<<(nx + 255) / 256, 256, 0, stream>>>(src, x, xbf, nx);
    }

    const dim3 blk(256);
    const dim3 gD(DD / 128, MM / 128);    // N=768
    const dim3 gF(DFFN / 128, MM / 128);  // N=3072
    const size_t attn_lds = (size_t)4 * 16 * COLS * 4 + (size_t)4 * 32 * 72 * 2;

    for (int l = 0; l < LL; ++l) {
        const unsigned short* qwl = qwb + (size_t)l * DD * DD;
        const unsigned short* kwl = kwb + (size_t)l * DD * DD;
        const unsigned short* vwl = vwb + (size_t)l * DD * DD;
        const unsigned short* owl = owb + (size_t)l * DD * DD;
        const unsigned short* w1l = w1b + (size_t)l * DFFN * DD;
        const unsigned short* w2l = w2b + (size_t)l * DD * DFFN;

        // QKV projections -> bf16
        gemm_bf16_kernel<<<gD, blk, 0, stream>>>(xbf, qwl, qb + l * DD, nullptr, qbf, MM, DD, DD, 0);
        gemm_bf16_kernel<<<gD, blk, 0, stream>>>(xbf, kwl, kb + l * DD, nullptr, kbf, MM, DD, DD, 0);
        gemm_bf16_kernel<<<gD, blk, 0, stream>>>(xbf, vwl, vb + l * DD, nullptr, vbf, MM, DD, DD, 0);

        // banded attention
        attn_kernel<<<(BB * HH * (SS / 16)) / 4, 128, attn_lds, stream>>>(qbf, kbf, vbf, ctx);

        // O projection -> f32, then residual+LN
        gemm_bf16_kernel<<<gD, blk, 0, stream>>>(ctx, owl, ob + l * DD, proj, nullptr, MM, DD, DD, 0);
        ln_kernel<<<MM, 256, 0, stream>>>(x, proj, g1 + l * DD, be1 + l * DD, x, xbf);

        // FFN
        gemm_bf16_kernel<<<gF, blk, 0, stream>>>(xbf, w1l, b1 + l * DFFN, nullptr, ffnbf, MM, DFFN, DD, 1);
        gemm_bf16_kernel<<<gD, blk, 0, stream>>>(ffnbf, w2l, b2 + l * DD, proj, nullptr, MM, DD, DFFN, 0);

        float* lnout = (l == LL - 1) ? (float*)d_out : x;
        ln_kernel<<<MM, 256, 0, stream>>>(x, proj, g2 + l * DD, be2 + l * DD, lnout, xbf);
    }
}
